// LinearSplineLayer_34574486733365
// MI455X (gfx1250) — compile-verified, matrix operand order checked by reading, not run on hardware
//
#include <hip/hip_runtime.h>
#include <math.h>
#include <stdint.h>

// ---------------------------------------------------------------------------
// LinearSplineLayer forward for MI455X (gfx1250, wave32, WMMA bf16)
//   B=2048, SIZE_IN=4096, SIZE_OUT=4096, K=8, HIDDEN=1024
// Pipeline:
//   stats -> normalize(bf16) -> transpose+cvt W1/W2 (bf16, [N][K] layout)
//   -> GEMM1 (tanh, bf16 out, 128x64 tile)
//   -> GEMM2 (128x128 tile) fused softmax/spline epilogue
//   -> log-density reduction
// GEMM staging uses CDNA5 GLOBAL_LOAD_ASYNC_TO_LDS_B128 (ASYNCcnt) into
// double-buffered LDS tiles: one workgroup barrier per K-step, copies fully
// overlapped with v_wmma_f32_16x16x32_bf16.
// ---------------------------------------------------------------------------

typedef __attribute__((ext_vector_type(16))) __bf16 v16bf;
typedef __attribute__((ext_vector_type(8)))  __bf16 v8bf;
typedef __attribute__((ext_vector_type(8)))  float  v8f;

#define B_ROWS   2048
#define SIZE_IN  4096
#define SIZE_OUT 4096
#define KBINS    8
#define HIDDEN   1024
#define N2TOT    (SIZE_OUT * KBINS)          // 32768
#define NPASS    (B_ROWS * SIZE_IN)          // 8388608
#define N2TILES  (N2TOT / 128)               // 256
#define EPSF     1e-6f
#define INV_PI   0.31830988618379067f

union FragBf { v16bf v; v8bf h[2]; };

// CDNA5 async global->LDS copy, 16 bytes per active lane (ASYNCcnt).
// vdst = LDS byte address (low 32 bits of generic pointer), vaddr = global.
__device__ __forceinline__ void async_copy_b128(const void* lds, const void* g) {
  uint32_t loff = (uint32_t)(uintptr_t)lds;
  uint64_t gaddr = (uint64_t)(uintptr_t)g;
  asm volatile("global_load_async_to_lds_b128 %0, %1, off"
               :: "v"(loff), "v"(gaddr) : "memory");
}
__device__ __forceinline__ void wait_async_all() {
  asm volatile("s_wait_asynccnt 0x0" ::: "memory");
}

// ---------------- stats: global mean & unbiased std of x_passive -----------
__global__ __launch_bounds__(256)
void stats_partial(const float* __restrict__ x, float* __restrict__ part, int n) {
  __shared__ float ssum[256], ssq[256];
  float s = 0.f, q = 0.f;
  for (int i = blockIdx.x * blockDim.x + threadIdx.x; i < n;
       i += gridDim.x * blockDim.x) {
    float v = x[i];
    s += v; q += v * v;
  }
  ssum[threadIdx.x] = s; ssq[threadIdx.x] = q;
  __syncthreads();
  for (int off = 128; off > 0; off >>= 1) {
    if ((int)threadIdx.x < off) {
      ssum[threadIdx.x] += ssum[threadIdx.x + off];
      ssq[threadIdx.x]  += ssq[threadIdx.x + off];
    }
    __syncthreads();
  }
  if (threadIdx.x == 0) {
    part[blockIdx.x]        = ssum[0];
    part[1024 + blockIdx.x] = ssq[0];
  }
}

__global__ __launch_bounds__(256)
void stats_finalize(const float* __restrict__ part, float* __restrict__ stats) {
  __shared__ float ssum[256], ssq[256];
  float s = 0.f, q = 0.f;
  for (int i = threadIdx.x; i < 1024; i += 256) {
    s += part[i]; q += part[1024 + i];
  }
  ssum[threadIdx.x] = s; ssq[threadIdx.x] = q;
  __syncthreads();
  for (int off = 128; off > 0; off >>= 1) {
    if ((int)threadIdx.x < off) {
      ssum[threadIdx.x] += ssum[threadIdx.x + off];
      ssq[threadIdx.x]  += ssq[threadIdx.x + off];
    }
    __syncthreads();
  }
  if (threadIdx.x == 0) {
    float n    = (float)NPASS;
    float mean = ssum[0] / n;
    float var  = (ssq[0] - ssum[0] * ssum[0] / n) / (n - 1.0f); // ddof=1
    stats[0] = mean;
    stats[1] = rsqrtf(var);
  }
}

// ---------------- normalize x_passive -> bf16 ------------------------------
__global__ __launch_bounds__(256)
void norm_to_bf16(const float* __restrict__ x, const float* __restrict__ stats,
                  __bf16* __restrict__ o, int n) {
  float mean = stats[0], inv = stats[1];
  for (int i = blockIdx.x * blockDim.x + threadIdx.x; i < n;
       i += gridDim.x * blockDim.x)
    o[i] = (__bf16)((x[i] - mean) * inv);
}

// ---------------- transpose + convert: W[K][N] f32 -> Wt[N][K] bf16 --------
__global__ __launch_bounds__(256)
void transpose_to_bf16(const float* __restrict__ W, __bf16* __restrict__ Wt,
                       int K, int N) {
  __shared__ float tile[32][33];
  const int t  = threadIdx.x;
  const int c  = t & 31;
  const int r0 = t >> 5;
  const int k0 = blockIdx.y * 32, n0 = blockIdx.x * 32;
#pragma unroll
  for (int i = 0; i < 4; ++i) {
    int r = r0 + 8 * i;
    tile[r][c] = W[(size_t)(k0 + r) * N + n0 + c];
  }
  __syncthreads();
#pragma unroll
  for (int i = 0; i < 4; ++i) {
    int r = r0 + 8 * i;
    Wt[(size_t)(n0 + r) * K + k0 + c] = (__bf16)tile[c][r];
  }
}

// ---------------- GEMM1: h = tanh(Xn @ W1 + b1) -> bf16 --------------------
// Block tile 128x64, 8 waves (4x2), wave tile 32x32; double-buffered LDS,
// async global->LDS staging.
__global__ __launch_bounds__(256)
void gemm1_tanh(const __bf16* __restrict__ A,   // [2048, 4096]
                const __bf16* __restrict__ Wt,  // [1024, 4096] (W1^T)
                const float*  __restrict__ bias,// [1024]
                __bf16* __restrict__ H) {       // [2048, 1024]
  __shared__ __align__(16) __bf16 As[2][128][32];
  __shared__ __align__(16) __bf16 Bs[2][64][32];
  const int t    = threadIdx.x;
  const int lane = t & 31, ll = lane & 15, lh = lane >> 4;
  const int wave = t >> 5, wm = wave >> 1, wn = wave & 1;
  const int bm = blockIdx.y * 128, bn = blockIdx.x * 64;
  const int K = SIZE_IN;
  const int colB = t >> 2, chB = (t & 3) * 8;
  v8f acc[2][2] = {};

  // prologue: stage buffer 0
#pragma unroll
  for (int i = 0; i < 2; ++i) {
    int id = t + 256 * i, row = id >> 2, ch = (id & 3) * 8;
    async_copy_b128(&As[0][row][ch], A + (size_t)(bm + row) * K + ch);
  }
  async_copy_b128(&Bs[0][colB][chB], Wt + (size_t)(bn + colB) * K + chB);
  wait_async_all();
  __syncthreads();

  for (int k0 = 0; k0 < K; k0 += 32) {
    const int cur = (k0 >> 5) & 1;
    if (k0 + 32 < K) {
#pragma unroll
      for (int i = 0; i < 2; ++i) {
        int id = t + 256 * i, row = id >> 2, ch = (id & 3) * 8;
        async_copy_b128(&As[cur ^ 1][row][ch],
                        A + (size_t)(bm + row) * K + k0 + 32 + ch);
      }
      async_copy_b128(&Bs[cur ^ 1][colB][chB],
                      Wt + (size_t)(bn + colB) * K + k0 + 32 + chB);
    }
    if (k0 + 64 < K) {
      __builtin_prefetch(A + (size_t)(bm + (t >> 1)) * K + k0 + 64, 0, 1);
      __builtin_prefetch(Wt + (size_t)(bn + colB) * K + k0 + 64 + chB, 0, 1);
    }
    FragBf af[2], bw[2];
#pragma unroll
    for (int mi = 0; mi < 2; ++mi) {
      const __bf16* p = &As[cur][wm * 32 + mi * 16 + ll][lh * 8];
      af[mi].h[0] = *(const v8bf*)p;            // K {0..7}/{8..15}
      af[mi].h[1] = *(const v8bf*)(p + 16);     // K {16..23}/{24..31}
    }
#pragma unroll
    for (int ni = 0; ni < 2; ++ni) {
      const __bf16* p = &Bs[cur][wn * 32 + ni * 16 + ll][lh * 8];
      bw[ni].h[0] = *(const v8bf*)p;
      bw[ni].h[1] = *(const v8bf*)(p + 16);
    }
#pragma unroll
    for (int mi = 0; mi < 2; ++mi)
#pragma unroll
      for (int ni = 0; ni < 2; ++ni)
        acc[mi][ni] = __builtin_amdgcn_wmma_f32_16x16x32_bf16(
            false, af[mi].v, false, bw[ni].v, (short)0, acc[mi][ni],
            false, false);
    wait_async_all();
    __syncthreads();
  }

#pragma unroll
  for (int mi = 0; mi < 2; ++mi)
#pragma unroll
    for (int ni = 0; ni < 2; ++ni) {
      int col = bn + wn * 32 + ni * 16 + ll;
      float bv = bias[col];
#pragma unroll
      for (int j = 0; j < 8; ++j) {
        int row = bm + wm * 32 + mi * 16 + lh * 8 + j;
        H[(size_t)row * HIDDEN + col] = (__bf16)tanhf(acc[mi][ni][j] + bv);
      }
    }
}

// ---------------- GEMM2 + fused softmax/spline epilogue --------------------
// Block tile 128x128, 8 waves (4x2), wave tile 32x64 (2x4 frags, 8 WMMA/step)
__global__ __launch_bounds__(256)
void gemm2_spline(const __bf16* __restrict__ Hm,  // [2048, 1024]
                  const __bf16* __restrict__ Wt,  // [32768, 1024] (W2^T)
                  const float*  __restrict__ bias,// [32768]
                  const float*  __restrict__ x_in,// [2048, 4096]
                  float* __restrict__ phi,        // [2048, 4096]
                  float* __restrict__ ldpart) {   // [256, 2048]
  __shared__ __align__(16) __bf16 As[2][128][32];
  __shared__ __align__(16) __bf16 Bs[2][128][32];
  __shared__ __align__(16) float  Ct[128][128];
  __shared__ float ldacc[128];
  const int t    = threadIdx.x;
  const int lane = t & 31, ll = lane & 15, lh = lane >> 4;
  const int wave = t >> 5, wm = wave >> 1, wn = wave & 1;
  const int bm = blockIdx.y * 128, bn = blockIdx.x * 128;
  const int K = HIDDEN;
  v8f acc[2][4] = {};

#pragma unroll
  for (int i = 0; i < 2; ++i) {
    int id = t + 256 * i, row = id >> 2, ch = (id & 3) * 8;
    async_copy_b128(&As[0][row][ch], Hm + (size_t)(bm + row) * K + ch);
    async_copy_b128(&Bs[0][row][ch], Wt + (size_t)(bn + row) * K + ch);
  }
  wait_async_all();
  __syncthreads();

  for (int k0 = 0; k0 < K; k0 += 32) {
    const int cur = (k0 >> 5) & 1;
    if (k0 + 32 < K) {
#pragma unroll
      for (int i = 0; i < 2; ++i) {
        int id = t + 256 * i, row = id >> 2, ch = (id & 3) * 8;
        async_copy_b128(&As[cur ^ 1][row][ch],
                        Hm + (size_t)(bm + row) * K + k0 + 32 + ch);
        async_copy_b128(&Bs[cur ^ 1][row][ch],
                        Wt + (size_t)(bn + row) * K + k0 + 32 + ch);
      }
    }
    if (k0 + 64 < K) {
      __builtin_prefetch(Hm + (size_t)(bm + (t >> 1)) * K + k0 + 64, 0, 1);
      __builtin_prefetch(Wt + (size_t)(bn + (t >> 1)) * K + k0 + 64, 0, 1);
    }
    FragBf af[2], bw[4];
#pragma unroll
    for (int mi = 0; mi < 2; ++mi) {
      const __bf16* p = &As[cur][wm * 32 + mi * 16 + ll][lh * 8];
      af[mi].h[0] = *(const v8bf*)p;
      af[mi].h[1] = *(const v8bf*)(p + 16);
    }
#pragma unroll
    for (int ni = 0; ni < 4; ++ni) {
      const __bf16* p = &Bs[cur][wn * 64 + ni * 16 + ll][lh * 8];
      bw[ni].h[0] = *(const v8bf*)p;
      bw[ni].h[1] = *(const v8bf*)(p + 16);
    }
#pragma unroll
    for (int mi = 0; mi < 2; ++mi)
#pragma unroll
      for (int ni = 0; ni < 4; ++ni)
        acc[mi][ni] = __builtin_amdgcn_wmma_f32_16x16x32_bf16(
            false, af[mi].v, false, bw[ni].v, (short)0, acc[mi][ni],
            false, false);
    wait_async_all();
    __syncthreads();
  }

  // tanh(net) tile -> LDS
#pragma unroll
  for (int mi = 0; mi < 2; ++mi)
#pragma unroll
    for (int ni = 0; ni < 4; ++ni) {
      int cl = wn * 64 + ni * 16 + ll;
      float bv = bias[bn + cl];
#pragma unroll
      for (int j = 0; j < 8; ++j) {
        int rl = wm * 32 + mi * 16 + lh * 8 + j;
        Ct[rl][cl] = tanhf(acc[mi][ni][j] + bv);
      }
    }
  if (t < 128) ldacc[t] = 0.f;
  __syncthreads();

  // per-(row, 8-bin group): softmax + linear-spline + log-density partial
  for (int task = t; task < 2048; task += 256) {
    int row = task >> 4, g = task & 15;
    float vals[8];
    *(float4*)&vals[0] = *(const float4*)&Ct[row][g * 8];
    *(float4*)&vals[4] = *(const float4*)&Ct[row][g * 8 + 4];
    float m = vals[0];
#pragma unroll
    for (int j = 1; j < 8; ++j) m = fmaxf(m, vals[j]);
    float e[8], sum = 0.f;
#pragma unroll
    for (int j = 0; j < 8; ++j) { e[j] = expf(vals[j] - m); sum += e[j]; }
    int rowg = bm + row;
    int sg   = blockIdx.x * 16 + g;
    float x  = x_in[(size_t)rowg * SIZE_OUT + sg] * INV_PI;
    // searchsorted(linspace(-eps,1+eps,9), x, 'left') - 1
    int ki = (int)ceilf((x + EPSF) * (8.0f / (1.0f + 2.0f * EPSF))) - 1;
    ki = min(7, max(0, ki));
    float pre = 0.f;
#pragma unroll
    for (int j = 0; j < 8; ++j) if (j < ki) pre += e[j];
    float inv   = 1.0f / sum;
    float pk    = e[ki] * inv;
    float alpha = (x - (float)ki * 0.125f) * 8.0f; // WIDTH = 1/K
    phi[(size_t)rowg * SIZE_OUT + sg] = pre * inv + alpha * pk;
    float logp = (vals[ki] - m) - logf(sum);
    atomicAdd(&ldacc[row], -logp);            // ds_add_f32, block-local
  }
  __syncthreads();
  if (t < 128)
    ldpart[(size_t)blockIdx.x * B_ROWS + bm + t] = ldacc[t];
}

// ---------------- final log-density reduction ------------------------------
__global__ __launch_bounds__(256)
void ld_final(const float* __restrict__ ldpart, const float* __restrict__ logd,
              float* __restrict__ out) {
  int b = blockIdx.x * blockDim.x + threadIdx.x;
  if (b >= B_ROWS) return;
  float acc = logd[b];
  for (int n = 0; n < N2TILES; ++n)
    acc += ldpart[(size_t)n * B_ROWS + b];
  out[b] = acc;
}

// ---------------------------------------------------------------------------
extern "C" void kernel_launch(void* const* d_in, const int* in_sizes, int n_in,
                              void* d_out, int out_size, void* d_ws, size_t ws_size,
                              hipStream_t stream) {
  (void)in_sizes; (void)n_in; (void)out_size; (void)ws_size;
  const float* x_in        = (const float*)d_in[0];
  const float* x_passive   = (const float*)d_in[1];
  const float* log_density = (const float*)d_in[2];
  const float* W1          = (const float*)d_in[3];
  const float* b1          = (const float*)d_in[4];
  const float* W2          = (const float*)d_in[5];
  const float* b2          = (const float*)d_in[6];

  float* out   = (float*)d_out;
  float* phi   = out;                               // [2048*4096]
  float* ldout = out + (size_t)B_ROWS * SIZE_OUT;   // [2048]

  char* ws = (char*)d_ws;   // workspace layout (~95 MB used)
  float*  part  = (float*)(ws + 0);                 //  8 KB  reduction partials
  float*  stats = (float*)(ws + 8192);              //  8 B   mean, invstd
  __bf16* xn    = (__bf16*)(ws + 8448ull);          // 16 MB  normalized x_passive
  __bf16* w1t   = (__bf16*)(ws + 16785664ull);      //  8 MB  W1^T bf16 [1024][4096]
  __bf16* w2t   = (__bf16*)(ws + 25174272ull);      // 64 MB  W2^T bf16 [32768][1024]
  __bf16* hb    = (__bf16*)(ws + 92283136ull);      //  4 MB  hidden bf16
  float*  ldp   = (float*)(ws + 96477440ull);       //  2 MB  [256][2048] partials

  stats_partial<<<1024, 256, 0, stream>>>(x_passive, part, NPASS);
  stats_finalize<<<1, 256, 0, stream>>>(part, stats);
  norm_to_bf16<<<4096, 256, 0, stream>>>(x_passive, stats, xn, NPASS);
  transpose_to_bf16<<<dim3(HIDDEN / 32, SIZE_IN / 32), 256, 0, stream>>>(
      W1, w1t, SIZE_IN, HIDDEN);
  transpose_to_bf16<<<dim3(N2TOT / 32, HIDDEN / 32), 256, 0, stream>>>(
      W2, w2t, HIDDEN, N2TOT);
  gemm1_tanh<<<dim3(HIDDEN / 64, B_ROWS / 128), 256, 0, stream>>>(
      xn, w1t, b1, hb);
  gemm2_spline<<<dim3(N2TILES, B_ROWS / 128), 256, 0, stream>>>(
      hb, w2t, b2, x_in, phi, ldp);
  ld_final<<<8, 256, 0, stream>>>(ldp, log_density, ldout);
}
